// SparseAttention_76304388981200
// MI455X (gfx1250) — matrix operand order repaired and checked by reading, not compiled, not run on hardware
//
#include <hip/hip_runtime.h>
#include <stdint.h>

#define DEV __device__ __forceinline__

typedef __attribute__((ext_vector_type(16))) __bf16 v16bf;
typedef __attribute__((ext_vector_type(8)))  float  v8f;

struct alignas(16) U4 { uint32_t a, b, c, d; };
struct alignas(8)  U2 { uint32_t a, b; };

union Frag {
  v16bf v;
  unsigned short s[16];
  U4 q[2];
};

DEV unsigned short f2bf(float f) {
  uint32_t u = __float_as_uint(f);
  u += 0x7FFFu + ((u >> 16) & 1u);
  return (unsigned short)(u >> 16);
}

DEV v8f wmma_bf16(const Frag& a, const Frag& b, v8f c) {
  return __builtin_amdgcn_wmma_f32_16x16x32_bf16(false, a.v, false, b.v,
                                                 (short)0, c, false, false);
}

// Async global -> LDS copy of 16 bytes per lane (CDNA5 ASYNCcnt path).
DEV void async_copy16(uint32_t lds_byte_off, uint64_t gaddr) {
  asm volatile("global_load_async_to_lds_b128 %0, %1, off"
               :: "v"(lds_byte_off), "v"(gaddr) : "memory");
}

// ---------------------------------------------------------------------------
// fp32 -> bf16 conversion (4 elems / thread)
// ---------------------------------------------------------------------------
__global__ __launch_bounds__(256)
void cvt_f32_bf16(const float* __restrict__ in, unsigned short* __restrict__ out, int n4) {
  int i = blockIdx.x * 256 + threadIdx.x;
  if (i < n4) {
    float4 f = ((const float4*)in)[i];
    U2 t;
    t.a = (uint32_t)f2bf(f.x) | ((uint32_t)f2bf(f.y) << 16);
    t.b = (uint32_t)f2bf(f.z) | ((uint32_t)f2bf(f.w) << 16);
    ((U2*)out)[i] = t;
  }
}

// ---------------------------------------------------------------------------
// bf16 WMMA GEMM with async double-buffered LDS staging:
//   y[m,n] = sum_k A[m,k] * Bw[n,k] + bias[n]
// A: [M,K] row-major bf16, Bw: [N,K] row-major bf16 (i.e. x @ W^T).
// Block = 8 waves; wave tile 32(M) x 64(N); block tile 128x128; k-step 32.
// Per k-step each wave issues 4 global_load_async_to_lds_b128 (A+B tiles),
// s_wait_asynccnt + barrier, 8 WMMAs read operands from LDS (80B padded rows
// -> bank-conflict-free 16-row operand reads), barrier, prefetch k+2.
// MODE 0: store bf16 per-head [B,H,S,Dk]            (Q, K)
// MODE 1: store bf16 per-head transposed [B,H,Dk,S] (V)
// MODE 3: store fp32 row-major [M,N]                (final output)
// ---------------------------------------------------------------------------
template<int MODE>
__global__ __launch_bounds__(256)
void gemm_bf16_k(const unsigned short* __restrict__ A,
                 const unsigned short* __restrict__ Bw,
                 const float* __restrict__ bias,
                 void* __restrict__ outv,
                 int M, int N, int K) {
  constexpr int ROWB  = 80;               // padded LDS row: 64B data + 16B pad
  constexpr int TILEH = 128 * ROWB / 2;   // 5120 halves per 128-row tile
  constexpr int BUFH  = 2 * TILEH;        // A tile + B tile per buffer

  __shared__ unsigned short gsm[2 * BUFH];   // 40 KB

  const int lane = threadIdx.x & 31;
  const int wave = threadIdx.x >> 5;
  const int wm = wave >> 1, wn = wave & 1;
  const int m_blk = blockIdx.y * 128;
  const int n_blk = blockIdx.x * 128;
  const int m_base = m_blk + wm * 32;
  const int n_base = n_blk + wn * 64;
  const int ln = lane & 15;
  const int hi = lane >> 4;

  const uint64_t Ag  = (uint64_t)(const char*)A;
  const uint64_t Bg  = (uint64_t)(const char*)Bw;
  const uint32_t sm0 = (uint32_t)(uintptr_t)(void*)gsm;
  const uint32_t Kb  = (uint32_t)K * 2;    // row stride in bytes

  // stage one 32-k slice of the 128-row A and B tiles (4 asyncs per wave)
  auto stage = [&](int st) {
    const uint32_t base  = sm0 + (uint32_t)((st & 1) * BUFH * 2);
    const uint32_t kbyte = (uint32_t)st * 64;
#pragma unroll
    for (int i = 0; i < 2; ++i) {
      uint32_t p   = (uint32_t)(wave * 2 + i) * 32 + lane;  // 0..511 pieces
      uint32_t row = p >> 2, col = (p & 3) * 16;            // 4 x 16B per row
      async_copy16(base + row * ROWB + col,
                   Ag + (uint64_t)(m_blk + row) * Kb + kbyte + col);
      async_copy16(base + (uint32_t)TILEH * 2 + row * ROWB + col,
                   Bg + (uint64_t)(n_blk + row) * Kb + kbyte + col);
    }
  };

  const v8f z8 = {0.f,0.f,0.f,0.f,0.f,0.f,0.f,0.f};
  v8f acc[2][4];
#pragma unroll
  for (int i = 0; i < 2; ++i)
#pragma unroll
    for (int j = 0; j < 4; ++j) acc[i][j] = z8;

  const int nst = K / 32;
  stage(0);
  stage(1);

  for (int st = 0; st < nst; ++st) {
    if (st + 1 < nst) asm volatile("s_wait_asynccnt 0x4" ::: "memory");
    else              asm volatile("s_wait_asynccnt 0x0" ::: "memory");
    __syncthreads();

    const unsigned short* At_ = gsm + (st & 1) * BUFH;
    const unsigned short* Bt_ = At_ + TILEH;

    Frag a0, a1, bfr[4];
    const unsigned short* ap0 = At_ + (size_t)(wm * 32 + ln) * (ROWB / 2);
    const unsigned short* ap1 = ap0 + 16 * (ROWB / 2);
    a0.q[0] = *(const U4*)(ap0 + hi * 8);
    a0.q[1] = *(const U4*)(ap0 + 16 + hi * 8);
    a1.q[0] = *(const U4*)(ap1 + hi * 8);
    a1.q[1] = *(const U4*)(ap1 + 16 + hi * 8);
#pragma unroll
    for (int j = 0; j < 4; ++j) {
      const unsigned short* bp = Bt_ + (size_t)(wn * 64 + j * 16 + ln) * (ROWB / 2);
      bfr[j].q[0] = *(const U4*)(bp + hi * 8);
      bfr[j].q[1] = *(const U4*)(bp + 16 + hi * 8);
    }
#pragma unroll
    for (int j = 0; j < 4; ++j) {
      acc[0][j] = wmma_bf16(a0, bfr[j], acc[0][j]);
      acc[1][j] = wmma_bf16(a1, bfr[j], acc[1][j]);
    }

    __syncthreads();
    if (st + 2 < nst) stage(st + 2);
  }

#pragma unroll
  for (int j = 0; j < 4; ++j) {
    const int n = n_base + j * 16 + ln;   // lane's output column
    const float bn = bias[n];
#pragma unroll
    for (int i = 0; i < 2; ++i) {
      const int m0 = m_base + i * 16 + hi * 8;  // first of 8 rows in this reg set
      if constexpr (MODE == 0) {
        unsigned short* o = (unsigned short*)outv;
        const int h = n >> 6, d = n & 63;
#pragma unroll
        for (int r = 0; r < 8; ++r) {
          const int m = m0 + r;
          const int bb = m >> 11, s = m & 2047;
          o[(((size_t)bb * 16 + h) * 2048 + s) * 64 + d] = f2bf(acc[i][j][r] + bn);
        }
      } else if constexpr (MODE == 1) {
        unsigned short* o = (unsigned short*)outv;
        const int h = n >> 6, d = n & 63;
        const int bb = m0 >> 11, s = m0 & 2047;   // same b for all 8 rows
        union { unsigned short s16[8]; U4 q; } pk;
#pragma unroll
        for (int r = 0; r < 8; ++r) pk.s16[r] = f2bf(acc[i][j][r] + bn);
        *(U4*)(o + (((size_t)bb * 16 + h) * 64 + d) * 2048 + s) = pk.q;
      } else {
        float* o = (float*)outv;
#pragma unroll
        for (int r = 0; r < 8; ++r)
          o[(size_t)(m0 + r) * N + n] = acc[i][j][r] + bn;
      }
    }
  }
}

// ---------------------------------------------------------------------------
// Sparse flash-attention with async double-buffered K/V staging in LDS.
// Workgroup = 8 waves, all sharing one (b,h); each wave owns 16 queries.
// Per 64-t chunk: each wave issues 4 global_load_async_to_lds_b128 (K + V^T),
// s_wait_asynccnt + barrier, then WMMAs read operands from LDS.
// Scores computed transposed (lane = s, regs = t) -> softmax needs only a
// half-wave shuffle and P feeds the PV WMMA as B-operand directly.
// ---------------------------------------------------------------------------
__global__ __launch_bounds__(256)
void sparse_attn_k(const unsigned short* __restrict__ Qh,
                   const unsigned short* __restrict__ Kh,
                   const unsigned short* __restrict__ Vt,
                   unsigned short* __restrict__ Attn) {
  constexpr int S = 2048, DK = 64;
  constexpr int CH = 64;                 // t-chunk size
  constexpr int NCH = S / CH;            // 32 chunks
  constexpr int ROWB = 144;              // padded LDS row: 128B data + 16B pad
  constexpr int TILEH = 64 * ROWB / 2;   // 4608 halves per 64-row tile
  constexpr int BUFH = 2 * TILEH;        // K tile + V tile per buffer

  __shared__ unsigned short smem[2 * BUFH];   // 36 KB

  const int lane = threadIdx.x & 31;
  const int wave = threadIdx.x >> 5;
  const int ln = lane & 15;
  const int hi = lane >> 4;
  const int bh = blockIdx.y;
  const int b  = bh >> 4, h = bh & 15;
  const int s0 = blockIdx.x * 128 + wave * 16;

  const size_t qkbase = (size_t)bh * S * DK;
  const size_t vbase  = (size_t)bh * DK * S;

  const uint64_t Kg  = (uint64_t)(const char*)(Kh + qkbase);  // [2048,64] bf16, rows 128B
  const uint64_t Vg  = (uint64_t)(const char*)(Vt + vbase);   // [64,2048] bf16, rows 4096B
  const uint32_t sm0 = (uint32_t)(uintptr_t)(void*)smem;      // LDS byte offset of smem

  // ---- Q B-operands, resident in registers for the whole t loop ----
  Frag qb0, qb1;
  {
    const unsigned short* qp = Qh + qkbase + (size_t)(s0 + ln) * DK + hi * 8;
    qb0.q[0] = *(const U4*)(qp);
    qb0.q[1] = *(const U4*)(qp + 16);
    qb1.q[0] = *(const U4*)(qp + 32);
    qb1.q[1] = *(const U4*)(qp + 48);
  }

  const int s_q = s0 + ln;      // this lane's query index
  const int tb  = hi * 8;       // own t offset within a 16-tile

  const v8f z8 = {0.f,0.f,0.f,0.f,0.f,0.f,0.f,0.f};
  float m_run = -1e30f, l_run = 0.f;
  v8f o[4];
#pragma unroll
  for (int dd = 0; dd < 4; ++dd) o[dd] = z8;

  // ---- async stage of one 64-t chunk: this wave's 4 pieces ----
  auto stage = [&](int c) {
    const uint32_t base = sm0 + (uint32_t)((c & 1) * BUFH * 2);
    const uint64_t kgc = Kg + (uint64_t)c * (CH * 128);   // contiguous 8 KB
    const uint64_t vgc = Vg + (uint64_t)c * (CH * 2);     // 64 strided rows of 128 B
#pragma unroll
    for (int inst = 0; inst < 2; ++inst) {
      // K: piece = 16B, 512 pieces, 8 per 128B row
      uint32_t g   = ((uint32_t)(wave * 2 + inst) * 32 + lane) * 16;
      uint32_t row = g >> 7, col = g & 127;
      async_copy16(base + row * ROWB + col, kgc + g);
      // V^T: same piece decomposition, but global rows are strided
      uint32_t p   = (uint32_t)(wave * 2 + inst) * 32 + lane;
      uint32_t vr  = p >> 3, vc = (p & 7) * 16;
      async_copy16(base + (uint32_t)TILEH * 2 + vr * ROWB + vc,
                   vgc + (uint64_t)vr * 4096 + vc);
    }
  };

  stage(0);
  stage(1);

  for (int c = 0; c < NCH; ++c) {
    if (c + 1 < NCH) asm volatile("s_wait_asynccnt 0x4" ::: "memory");
    else             asm volatile("s_wait_asynccnt 0x0" ::: "memory");
    __syncthreads();

    const unsigned short* kt = smem + (c & 1) * BUFH;   // rows: t - c*64, stride 72 halves
    const unsigned short* vt = kt + TILEH;              // rows: d,        stride 72 halves

#pragma unroll
    for (int half = 0; half < 2; ++half) {
      const int ts = half * 32;
      // ---- S^T tiles from LDS: st1 = t ts..ts+15, st2 = ts+16..ts+31 ----
      v8f st1 = z8, st2 = z8;
      Frag ka;
      const unsigned short* kp1 = kt + (size_t)(ts + ln) * (ROWB / 2);
      const unsigned short* kp2 = kp1 + 16 * (ROWB / 2);
      ka.q[0] = *(const U4*)(kp1 + tb);       ka.q[1] = *(const U4*)(kp1 + 16 + tb);
      st1 = wmma_bf16(ka, qb0, st1);
      ka.q[0] = *(const U4*)(kp1 + 32 + tb);  ka.q[1] = *(const U4*)(kp1 + 48 + tb);
      st1 = wmma_bf16(ka, qb1, st1);
      ka.q[0] = *(const U4*)(kp2 + tb);       ka.q[1] = *(const U4*)(kp2 + 16 + tb);
      st2 = wmma_bf16(ka, qb0, st2);
      ka.q[0] = *(const U4*)(kp2 + 32 + tb);  ka.q[1] = *(const U4*)(kp2 + 48 + tb);
      st2 = wmma_bf16(ka, qb1, st2);

      // ---- scale + sparse mask + local max ----
      const int tA = c * CH + ts + tb;
      float mloc = -1e30f;
#pragma unroll
      for (int r = 0; r < 8; ++r) {
        const int t1 = tA + r;
        const int t2 = t1 + 16;
        float v1 = st1[r] * 0.125f;
        float v2 = st2[r] * 0.125f;
        int a1 = t1 - s_q; if (a1 < 0) a1 = -a1;
        int a2 = t2 - s_q; if (a2 < 0) a2 = -a2;
        const bool ok1 = (a1 <= 32) || (((t1 ^ s_q) & 3) == 0);
        const bool ok2 = (a2 <= 32) || (((t2 ^ s_q) & 3) == 0);
        v1 = ok1 ? v1 : -1e30f;
        v2 = ok2 ? v2 : -1e30f;
        st1[r] = v1; st2[r] = v2;
        mloc = fmaxf(mloc, fmaxf(v1, v2));
      }
      mloc = fmaxf(mloc, __shfl_xor(mloc, 16));  // pair lane: same s, other t half
      const float mnew  = fmaxf(m_run, mloc);    // every 32-t chunk has live cols
      const float alpha = __expf(m_run - mnew);
      l_run *= alpha;
#pragma unroll
      for (int dd = 0; dd < 4; ++dd)
#pragma unroll
        for (int r = 0; r < 8; ++r) o[dd][r] *= alpha;

      float lloc = 0.f;
#pragma unroll
      for (int r = 0; r < 8; ++r) {
        const float p1 = __expf(st1[r] - mnew);
        const float p2 = __expf(st2[r] - mnew);
        st1[r] = p1; st2[r] = p2;
        lloc += p1 + p2;
      }
      l_run += lloc;
      m_run = mnew;

      // ---- build P as WMMA B-operand ----
      Frag pb;
#pragma unroll
      for (int r = 0; r < 8; ++r) {
        const float p1o = __shfl_xor(st1[r], 16);
        const float p2o = __shfl_xor(st2[r], 16);
        pb.s[r]     = f2bf(hi ? p2o    : st1[r]);
        pb.s[8 + r] = f2bf(hi ? st2[r] : p1o);
      }

      // ---- Out^T[d, s] += V^T[d, t] * P[t, s] (A operand from LDS) ----
#pragma unroll
      for (int dd = 0; dd < 4; ++dd) {
        Frag va;
        const unsigned short* vp = vt + (size_t)(dd * 16 + ln) * (ROWB / 2) + ts + hi * 16;
        va.q[0] = *(const U4*)(vp);
        va.q[1] = *(const U4*)(vp + 8);
        o[dd] = wmma_bf16(va, pb, o[dd]);
      }
    }

    __syncthreads();                 // all waves done reading this buffer
    if (c + 2 < NCH) stage(c + 2);   // refill it for chunk c+2
  }

  // ---- normalize and store bf16 to Attn[b, s, h*64 + d] ----
  const float ltot = l_run + __shfl_xor(l_run, 16);
  const float inv = 1.0f / ltot;
  const size_t rowbase = ((size_t)b * S + s_q) * 1024 + h * 64 + hi * 8;
#pragma unroll
  for (int dd = 0; dd < 4; ++dd) {
    union { unsigned short s16[8]; U4 q; } pk;
#pragma unroll
    for (int r = 0; r < 8; ++r) pk.s16[r] = f2bf(o[dd][r] * inv);
    *(U4*)(Attn + rowbase + dd * 16) = pk.q;
  }
}

// ---------------------------------------------------------------------------
extern "C" void kernel_launch(void* const* d_in, const int* in_sizes, int n_in,
                              void* d_out, int out_size, void* d_ws, size_t ws_size,
                              hipStream_t stream) {
  (void)in_sizes; (void)n_in; (void)out_size; (void)ws_size;
  const float* x  = (const float*)d_in[0];
  const float* Wq = (const float*)d_in[1];
  const float* bq = (const float*)d_in[2];
  const float* Wk = (const float*)d_in[3];
  const float* bk = (const float*)d_in[4];
  const float* Wv = (const float*)d_in[5];
  const float* bv = (const float*)d_in[6];
  const float* Wo = (const float*)d_in[7];
  const float* bo = (const float*)d_in[8];
  float* out = (float*)d_out;

  constexpr size_t XE = (size_t)4096 * 1024;   // x / Q / K / V / Attn elements
  constexpr size_t WE = (size_t)1024 * 1024;   // weight elements
  unsigned short* ws  = (unsigned short*)d_ws;
  unsigned short* xbf = ws;
  unsigned short* wqb = xbf + XE;
  unsigned short* wkb = wqb + WE;
  unsigned short* wvb = wkb + WE;
  unsigned short* wob = wvb + WE;
  unsigned short* Qh  = wob + WE;
  unsigned short* Kh  = Qh + XE;
  unsigned short* Vt  = Kh + XE;
  unsigned short* At  = Vt + XE;

  cvt_f32_bf16<<<dim3((unsigned)(XE / 4 / 256)), 256, 0, stream>>>(x,  xbf, (int)(XE / 4));
  cvt_f32_bf16<<<dim3((unsigned)(WE / 4 / 256)), 256, 0, stream>>>(Wq, wqb, (int)(WE / 4));
  cvt_f32_bf16<<<dim3((unsigned)(WE / 4 / 256)), 256, 0, stream>>>(Wk, wkb, (int)(WE / 4));
  cvt_f32_bf16<<<dim3((unsigned)(WE / 4 / 256)), 256, 0, stream>>>(Wv, wvb, (int)(WE / 4));
  cvt_f32_bf16<<<dim3((unsigned)(WE / 4 / 256)), 256, 0, stream>>>(Wo, wob, (int)(WE / 4));

  dim3 gg(8, 32);  // N/128, M/128
  gemm_bf16_k<0><<<gg, 256, 0, stream>>>(xbf, wqb, bq, (void*)Qh, 4096, 1024, 1024);
  gemm_bf16_k<0><<<gg, 256, 0, stream>>>(xbf, wkb, bk, (void*)Kh, 4096, 1024, 1024);
  gemm_bf16_k<1><<<gg, 256, 0, stream>>>(xbf, wvb, bv, (void*)Vt, 4096, 1024, 1024);

  sparse_attn_k<<<dim3(16, 32), 256, 0, stream>>>(Qh, Kh, Vt, At);

  gemm_bf16_k<3><<<gg, 256, 0, stream>>>(At, wob, bo, (void*)out, 4096, 1024, 1024);
}
